// GFBc_57509612093670
// MI455X (gfx1250) — matrix operand order repaired and checked by reading, not compiled, not run on hardware
//
#include <hip/hip_runtime.h>
#include <hip/hip_bf16.h>
#include <stdint.h>

typedef __attribute__((ext_vector_type(16))) _Float16 v16h;
typedef __attribute__((ext_vector_type(8)))  float    v8f;

#define K_RAW   625          // 25*25
#define KROWS   25
#define KCOLS   32           // filter row padded 25 -> 32
#define K_PAD   (KROWS*KCOLS)   // 800
#define OCH     512
#define IMG_HW  224
#define OUT_HW  56
#define OUT_P   (OUT_HW*OUT_HW)   // 3136
#define CSTRIDE 4
#define CPAD    12
#define KSZ     25

// LDS input tile: a 64-position block spans <=2 output rows (blocks step by
// 64 = 8 mod 56, so p%56 <= 48 -> one row boundary max).
// Rows: 25 (filter) + 4 (stride to 2nd row) + 1 margin = 30.
// Cols: 224 + 2*12 halo = 248.  (kx overread <=3 stays inside the buffer and
// is multiplied by zero-padded weights.)
#define TILE_ROWS 30
#define TILE_COLS 248

// ---- pass 1: pack gabor bank f32 -> f16, row-pad kx 25 -> 32 ---------------
__global__ void gabor_w2h(const float* __restrict__ w, _Float16* __restrict__ wf16) {
    int idx = blockIdx.x * blockDim.x + threadIdx.x;          // o*800 + ky*32 + kx
    if (idx >= OCH * K_PAD) return;
    int o   = idx / K_PAD;
    int kr  = idx - o * K_PAD;
    int ky  = kr / KCOLS;
    int kx  = kr - ky * KCOLS;
    float v = (kx < KSZ) ? w[o * K_RAW + ky * KSZ + kx] : 0.0f;
    wf16[idx] = (_Float16)v;
}

// ---- pass 2: implicit-GEMM conv via v_wmma_f32_16x16x32_f16 ----------------
// Wave: 64 channels (4 M tiles) x 32 positions (2 N slices) = 8 accumulators.
// Block: 512 threads = 16 waves (8 ch-groups x 2 N-groups)
//        -> 512 channels x 64 positions. One K chunk = one filter row.
__global__ __launch_bounds__(512)
void gabor_wmma(const float* __restrict__ x, const _Float16* __restrict__ wf16,
                float* __restrict__ out) {
    __shared__ _Float16 tile[TILE_ROWS * TILE_COLS];   // 14.9 KB

    const int lane   = threadIdx.x & 31;
    const int wave   = threadIdx.x >> 5;
    const int mgrp   = wave & 7;                       // 64-channel group
    const int ngrp   = wave >> 3;                      // 32-position group
    const int img    = blockIdx.y;                     // 0..47
    const int pblk   = blockIdx.x * 64;                // first output position

    const float* xb = x + (size_t)img * IMG_HW * IMG_HW;

    // ---- stage zero-padded input tile to LDS (coalesced, once) ----
    const int oy0 = pblk / OUT_HW;
    const int iyT = oy0 * CSTRIDE - CPAD;              // iy of tile row 0
    for (int i = threadIdx.x; i < TILE_ROWS * TILE_COLS; i += 512) {
        int r  = i / TILE_COLS;
        int c  = i - r * TILE_COLS;
        int iy = iyT + r;
        int ix = c - CPAD;
        float v = 0.0f;
        if ((unsigned)iy < IMG_HW && (unsigned)ix < IMG_HW)
            v = xb[iy * IMG_HW + ix];
        tile[i] = (_Float16)v;
    }
    __syncthreads();

    const int ncol = lane & 15;
    const int half = lane >> 4;

    // per-N-slice output position and im2col origin inside the LDS tile
    int pS[2], ldsIdx[2];
    #pragma unroll
    for (int s = 0; s < 2; ++s) {
        int p  = pblk + ngrp * 32 + s * 16 + ncol;
        int oy = p / OUT_HW;
        int ox = p - oy * OUT_HW;
        pS[s]     = p;
        ldsIdx[s] = (oy - oy0) * CSTRIDE * TILE_COLS + ox * CSTRIDE + half * 16;
    }

    v8f acc[2][4];
    #pragma unroll
    for (int s = 0; s < 2; ++s)
        #pragma unroll
        for (int t = 0; t < 4; ++t) acc[s][t] = (v8f){0,0,0,0,0,0,0,0};

    // A-matrix row this lane holds (channel), K laid out as ky*32 + kx
    const _Float16* wr  = wf16 + (size_t)(mgrp * 64 + ncol) * K_PAD + half * 16;
    const _Float16* tb0 = tile + ldsIdx[0];
    const _Float16* tb1 = tile + ldsIdx[1];

    #pragma unroll 1
    for (int ky = 0; ky < KROWS; ++ky) {
        // ---- A fragments: 8 aligned b128 loads (one clause) ----
        union { v16h h; uint4 q[2]; } af[4];
        #pragma unroll
        for (int t = 0; t < 4; ++t) {
            af[t].q[0] = *(const uint4*)(wr + (size_t)t * 16 * K_PAD);
            af[t].q[1] = *(const uint4*)(wr + (size_t)t * 16 * K_PAD + 8);
        }
        __builtin_prefetch(wr + KCOLS, 0, 3);

        // ---- B fragments: 16 contiguous f16 per slice -> ds_load_2addr_b64 ----
        union { v16h h; uint2 d[4]; } bf[2];
        #pragma unroll
        for (int j = 0; j < 4; ++j) bf[0].d[j] = *(const uint2*)(tb0 + j * 4);
        #pragma unroll
        for (int j = 0; j < 4; ++j) bf[1].d[j] = *(const uint2*)(tb1 + j * 4);

        // ---- 8 back-to-back WMMAs on independent accumulators ----
        #pragma unroll
        for (int s = 0; s < 2; ++s)
            #pragma unroll
            for (int t = 0; t < 4; ++t)
                acc[s][t] = __builtin_amdgcn_wmma_f32_16x16x32_f16(
                    false, af[t].h, false, bf[s].h, (short)0, acc[s][t],
                    false, false);

        wr  += KCOLS;
        tb0 += TILE_COLS;
        tb1 += TILE_COLS;
    }

    // ---- D tiles: VGPR r, lane l -> channel (base + r + 8*half), col l%16 ----
    float* ob = out + (size_t)img * OCH * OUT_P;
    #pragma unroll
    for (int s = 0; s < 2; ++s) {
        float* obs = ob + pS[s];
        #pragma unroll
        for (int t = 0; t < 4; ++t) {
            int obase = mgrp * 64 + t * 16 + half * 8;
            #pragma unroll
            for (int r = 0; r < 8; ++r)
                obs[(size_t)(obase + r) * OUT_P] = acc[s][t][r];
        }
    }
}

extern "C" void kernel_launch(void* const* d_in, const int* in_sizes, int n_in,
                              void* d_out, int out_size, void* d_ws, size_t ws_size,
                              hipStream_t stream) {
    const float* x = (const float*)d_in[0];   // [16,3,224,224] -> 48 x 224x224
    const float* w = (const float*)d_in[1];   // [512,1,25,25]
    float* out = (float*)d_out;               // [16,3,512,56,56]
    _Float16* wf16 = (_Float16*)d_ws;         // 512*800 f16 = 800 KB scratch

    const int npack = OCH * K_PAD;
    gabor_w2h<<<(npack + 255) / 256, 256, 0, stream>>>(w, wf16);

    dim3 grid(OUT_P / 64, 48);                // 49 x 48 blocks, 512 thr each
    gabor_wmma<<<grid, 512, 0, stream>>>(x, wf16, out);
}
